// BEE_Bin2Symbol_28243704938870
// MI455X (gfx1250) — matrix operand order repaired and checked by reading, not compiled, not run on hardware
//
#include <hip/hip_runtime.h>
#include <hip/hip_bf16.h>

typedef _Float16 half_t;
typedef __attribute__((ext_vector_type(16))) _Float16 v16h;
typedef __attribute__((ext_vector_type(8)))  _Float16 v8h;
typedef __attribute__((ext_vector_type(8)))  float    v8f;

#define MCH   192
#define HH    32
#define WW    48
#define HW    (HH*WW)
#define PH    36
#define PW    52
#define PHW   (PH*PW)
#define KCTX  2304          // 192 ch * 12 causal taps
#define NT    512
#define NWAVE 16
// LDS row strides (halves), 16B-aligned rows
#define CS  2320            // 2304 + 16
#define S0  784             // 768 + 16
#define S1  656             // 640 + 16

// ---------------- hyper-decoder helper kernels (fp32, parallel) -------------

__global__ void hd_deconv5s2_lrelu(const float* __restrict__ in, const float* __restrict__ w,
                                   const float* __restrict__ b, float* __restrict__ out,
                                   int CIN, int COUT, int IH, int IW) {
  int OH = IH * 2, OW = IW * 2;
  int idx = blockIdx.x * blockDim.x + threadIdx.x;
  if (idx >= COUT * OH * OW) return;
  int x = idx % OW, y = (idx / OW) % OH, o = idx / (OW * OH);
  float acc = b[o];
  for (int kh = 0; kh < 5; ++kh) {
    int ty = y + 2 - kh; if (ty & 1) continue; int iy = ty >> 1;
    if ((unsigned)iy >= (unsigned)IH) continue;
    for (int kw = 0; kw < 5; ++kw) {
      int tx = x + 2 - kw; if (tx & 1) continue; int ix = tx >> 1;
      if ((unsigned)ix >= (unsigned)IW) continue;
      const float* ip = in + iy * IW + ix;
      const float* wp = w + o * 25 + kh * 5 + kw;     // w layout [CIN][COUT][5][5]
      for (int c = 0; c < CIN; ++c)
        acc = fmaf(ip[c * IH * IW], wp[c * COUT * 25], acc);
    }
  }
  out[idx] = acc > 0.f ? acc : 0.01f * acc;
}

__global__ void hd_conv3x3(const float* __restrict__ in, const float* __restrict__ w,
                           const float* __restrict__ b, float* __restrict__ out,
                           int CIN, int COUT, int H, int W) {
  int idx = blockIdx.x * blockDim.x + threadIdx.x;
  if (idx >= COUT * H * W) return;
  int x = idx % W, y = (idx / W) % H, o = idx / (W * H);
  float acc = b[o];
  for (int c = 0; c < CIN; ++c) {
    const float* ip = in + c * H * W;
    const float* wp = w + (o * CIN + c) * 9;          // [COUT][CIN][3][3]
    for (int dy = 0; dy < 3; ++dy) {
      int yy = y + dy - 1; if ((unsigned)yy >= (unsigned)H) continue;
      for (int dx = 0; dx < 3; ++dx) {
        int xx = x + dx - 1; if ((unsigned)xx >= (unsigned)W) continue;
        acc = fmaf(ip[yy * W + xx], wp[dy * 3 + dx], acc);
      }
    }
  }
  out[idx] = acc;
}

// ---------------- weight packing into WMMA B-fragment layout ----------------
// flat index e: block = e>>9 (32 lanes * 16 halves), lane holds column
// N = ct*16 + (lane&15), K halves koff..koff+15 with koff = (lane<16 ? 0 : 16).

__global__ void pack_mlp_w(const float* __restrict__ W, half_t* __restrict__ dst,
                           int N, int K) {
  int e = blockIdx.x * blockDim.x + threadIdx.x;
  if (e >= N * K) return;
  int blk = e >> 9, rem = e & 511;
  int lane = rem >> 4, h = rem & 15;
  int nks = K >> 5;
  int ct = blk / nks, ks = blk - ct * nks;
  int n = ct * 16 + (lane & 15);
  int k = ks * 32 + ((lane < 16) ? 0 : 16) + h;
  dst[e] = (half_t)W[n * K + k];
}

__global__ void pack_ctx_w(const float* __restrict__ W, half_t* __restrict__ dst) {
  const int N = 2 * MCH, K = KCTX;
  int e = blockIdx.x * blockDim.x + threadIdx.x;
  if (e >= N * K) return;
  int blk = e >> 9, rem = e & 511;
  int lane = rem >> 4, h = rem & 15;
  int nks = K >> 5;                                    // 72
  int ct = blk / nks, ks = blk - ct * nks;
  int n = ct * 16 + (lane & 15);
  int k = ks * 32 + ((lane < 16) ? 0 : 16) + h;
  int c = k / 12, q = k - c * 12;                      // 12 causal taps (type-A mask)
  int dy = (q < 5) ? 0 : ((q < 10) ? 1 : 2);
  int dx = (q < 5) ? q : ((q < 10) ? (q - 5) : (q - 10));
  dst[e] = (half_t)W[((n * MCH + c) * 5 + dy) * 5 + dx];  // ctx_w [2M][M][5][5]
}

__global__ void zero_half_buf(half_t* __restrict__ p, int n) {
  int i = blockIdx.x * blockDim.x + threadIdx.x;
  if (i < n) p[i] = (half_t)0.f;
}

// ---------------- WMMA tile accumulator -------------------------------------
// A (16 x 32k) streamed from LDS with lane half h holding k 16h..16h+15, B from packed global.

__device__ __forceinline__ v8f wmma_accum(const half_t* __restrict__ Xl, int xstride,
                                          const half_t* __restrict__ Wb, int nks, int lane) {
  v8f acc = {};
  int row = lane & 15;
  int kb  = (lane < 16) ? 0 : 16;
  const half_t* xr = Xl + row * xstride + kb;
  for (int ks = 0; ks < nks; ++ks) {
    union { v16h v; v8h h[2]; } a, b;
    a.h[0] = *(const v8h*)(xr + ks * 32);
    a.h[1] = *(const v8h*)(xr + ks * 32 + 8);
    const half_t* bp = Wb + ks * 512 + lane * 16;
    b.h[0] = *(const v8h*)(bp);
    b.h[1] = *(const v8h*)(bp + 8);
    __builtin_prefetch(bp + 2048, 0, 1);               // 4 k-steps ahead -> global_prefetch
    acc = __builtin_amdgcn_wmma_f32_16x16x32_f16(false, a.v, false, b.v,
                                                 (short)0, acc, false, false);
  }
  return acc;
}

// ---------------- persistent autoregressive wavefront kernel ----------------

__launch_bounds__(NT, 1)
__global__ void ar_decode_wavefront(
    const float* __restrict__ fm1, const float* __restrict__ w_hat,
    const half_t* __restrict__ wctx, const half_t* __restrict__ wmlp,
    const float* __restrict__ ctx_b,
    const float* __restrict__ b0, const float* __restrict__ b1,
    const float* __restrict__ b2, const float* __restrict__ b3,
    const float* __restrict__ b4, const float* __restrict__ b5,
    half_t* __restrict__ sQLh, float* __restrict__ out) {
  extern __shared__ char smem[];
  half_t* Xctx  = (half_t*)smem;               // 16 * CS halves
  half_t* X0    = Xctx + 16 * CS;              // 16 * S0
  half_t* X1    = X0 + 16 * S0;                // 16 * S1
  float*  stage = (float*)(X1 + 16 * S1);      // 16 * 384 fp32 (async landing zone)

  const int tid = threadIdx.x, lane = tid & 31, wave = tid >> 5;
  const int dims[7] = {768, 640, 512, 384, 320, 256, 192};
  int woff[6]; { int a = 0; for (int l = 0; l < 6; ++l) { woff[l] = a; a += dims[l] * dims[l + 1]; } }
  const float* bias[6] = {b0, b1, b2, b3, b4, b5};

  for (int t = 0; t < 3 * (HH - 1) + WW; ++t) {        // 141 wavefronts, t = 3i + j
    int imin = (t >= WW) ? (t - (WW - 1) + 2) / 3 : 0;
    int imax = t / 3; if (imax > HH - 1) imax = HH - 1;
    int count = imax - imin + 1;                       // <= 16

    // (1) gather causal-mask patches -> Xctx (f16)
    for (int e = tid; e < 16 * KCTX; e += NT) {
      int p = e / KCTX, k = e - p * KCTX;
      half_t v = (half_t)0.f;
      if (p < count) {
        int i = imin + p, j = t - 3 * i;
        int c = k / 12, q = k - c * 12;
        int dy = (q < 5) ? 0 : ((q < 10) ? 1 : 2);
        int dx = (q < 5) ? q : ((q < 10) ? (q - 5) : (q - 10));
        v = sQLh[c * PHW + (i + dy) * PW + (j + dx)];
      }
      Xctx[p * CS + k] = v;
    }

    // (2) async gather of fm1 feature columns straight into LDS
    for (int e = tid; e < 16 * 2 * MCH; e += NT) {
      int p = e / (2 * MCH), c = e - p * 2 * MCH;
      if (p < count) {
        int i = imin + p, j = t - 3 * i;
        unsigned laddr = (unsigned)(size_t)(stage + e);
        unsigned long long gaddr = (unsigned long long)(size_t)(fm1 + c * HW + i * WW + j);
        asm volatile("global_load_async_to_lds_b32 %0, %1, off"
                     :: "v"(laddr), "v"(gaddr) : "memory");
      }
    }
    asm volatile("s_wait_asynccnt 0" ::: "memory");
    __syncthreads();

    // (3a) convert staged fm1 -> X0[:, 0:384]
    for (int e = tid; e < 16 * 2 * MCH; e += NT) {
      int p = e / (2 * MCH), c = e - p * 2 * MCH;
      X0[p * S0 + c] = (p < count) ? (half_t)stage[e] : (half_t)0.f;
    }
    // (3b) masked-conv context GEMM: [16 x 2304] x [2304 x 384] -> X0[:, 384:768]
    for (int ct = wave; ct < (2 * MCH) / 16; ct += NWAVE) {
      v8f acc = wmma_accum(Xctx, CS, wctx + ct * (KCTX / 32) * 512, KCTX / 32, lane);
      int n = (lane & 15) + ct * 16;
      float bb = ctx_b[n];
      int mo = (lane < 16) ? 0 : 8;
#pragma unroll
      for (int r = 0; r < 8; ++r)
        X0[(r + mo) * S0 + 2 * MCH + n] = (half_t)(acc[r] + bb);
    }
    __syncthreads();

    // (4) entropy-parameter MLP, 6 layers, ping-pong LDS buffers
    for (int l = 0; l < 6; ++l) {
      half_t* Xi = (l & 1) ? X1 : X0;
      half_t* Xo = (l & 1) ? X0 : X1;
      int Si = (l & 1) ? S1 : S0;
      int So = (l & 1) ? S0 : S1;
      int nks = dims[l] >> 5, ntl = dims[l + 1] >> 4;
      const half_t* wl = wmlp + woff[l];
      for (int ct = wave; ct < ntl; ct += NWAVE) {
        v8f acc = wmma_accum(Xi, Si, wl + ct * nks * 512, nks, lane);
        int n = (lane & 15) + ct * 16;
        float bb = bias[l][n];
        int mo = (lane < 16) ? 0 : 8;
        if (l < 5) {
#pragma unroll
          for (int r = 0; r < 8; ++r) {
            float v = acc[r] + bb;
            v = v > 0.f ? v : 0.01f * v;               // leaky relu
            Xo[(r + mo) * So + n] = (half_t)v;
          }
        } else {                                       // final: +residual, commit pixel
#pragma unroll
          for (int r = 0; r < 8; ++r) {
            int slot = r + mo;
            if (slot < count) {
              int i = imin + slot, j = t - 3 * i;
              float v = acc[r] + bb + w_hat[n * HW + i * WW + j];
              out[n * HW + i * WW + j] = v;
              sQLh[n * PHW + (i + 2) * PW + (j + 2)] = (half_t)v;
            }
          }
        }
      }
      __syncthreads();
    }
    __threadfence();                                   // sQLh visible to next wavefront
    __builtin_amdgcn_s_cluster_barrier();              // NOP when unclustered
    __syncthreads();
  }
}

// ---------------- launch ----------------------------------------------------

extern "C" void kernel_launch(void* const* d_in, const int* in_sizes, int n_in,
                              void* d_out, int out_size, void* d_ws, size_t ws_size,
                              hipStream_t stream) {
  (void)in_sizes; (void)n_in; (void)out_size; (void)ws_size;
  const float* z_hat  = (const float*)d_in[0];
  const float* w_hat  = (const float*)d_in[1];
  const float* hs_dw0 = (const float*)d_in[2];
  const float* hs_db0 = (const float*)d_in[3];
  const float* hs_dw1 = (const float*)d_in[4];
  const float* hs_db1 = (const float*)d_in[5];
  const float* hs_cw2 = (const float*)d_in[6];
  const float* hs_cb2 = (const float*)d_in[7];
  const float* ctx_w  = (const float*)d_in[8];
  const float* ctx_b  = (const float*)d_in[9];
  const float* ep_w[6]; const float* ep_b[6];
  for (int l = 0; l < 6; ++l) { ep_w[l] = (const float*)d_in[10 + 2 * l];
                                ep_b[l] = (const float*)d_in[11 + 2 * l]; }

  char* ws = (char*)d_ws; size_t off = 0;
  auto carve = [&](size_t bytes) { void* p = ws + off; off += (bytes + 255) & ~(size_t)255; return p; };
  float*  t0   = (float*)carve((size_t)MCH * 16 * 24 * 4);
  float*  t1   = (float*)carve((size_t)(3 * MCH / 2) * HH * WW * 4);
  float*  fm1  = (float*)carve((size_t)(2 * MCH) * HH * WW * 4);
  half_t* sQLh = (half_t*)carve((size_t)MCH * PHW * 2);
  half_t* wctx = (half_t*)carve((size_t)KCTX * 2 * MCH * 2);
  const int dims[7] = {768, 640, 512, 384, 320, 256, 192};
  size_t mlpHalves = 0; for (int l = 0; l < 6; ++l) mlpHalves += (size_t)dims[l] * dims[l + 1];
  half_t* wmlp = (half_t*)carve(mlpHalves * 2);

  // weight packing + sQL clear (independent of conv stack)
  { int n = KCTX * 2 * MCH;
    pack_ctx_w<<<(n + 255) / 256, 256, 0, stream>>>(ctx_w, wctx); }
  { size_t a = 0;
    for (int l = 0; l < 6; ++l) {
      int n = dims[l] * dims[l + 1];
      pack_mlp_w<<<(n + 255) / 256, 256, 0, stream>>>(ep_w[l], wmlp + a, dims[l + 1], dims[l]);
      a += (size_t)n;
    } }
  { int n = MCH * PHW;
    zero_half_buf<<<(n + 255) / 256, 256, 0, stream>>>(sQLh, n); }

  // hyper-decoder: deconv -> deconv -> conv (lrelu fused into deconv outputs)
  { int n = MCH * 16 * 24;
    hd_deconv5s2_lrelu<<<(n + 255) / 256, 256, 0, stream>>>(z_hat, hs_dw0, hs_db0, t0,
                                                            MCH, MCH, 8, 12); }
  { int n = (3 * MCH / 2) * HH * WW;
    hd_deconv5s2_lrelu<<<(n + 255) / 256, 256, 0, stream>>>(t0, hs_dw1, hs_db1, t1,
                                                            MCH, 3 * MCH / 2, 16, 24); }
  { int n = (2 * MCH) * HH * WW;
    hd_conv3x3<<<(n + 255) / 256, 256, 0, stream>>>(t1, hs_cw2, hs_cb2, fm1,
                                                    3 * MCH / 2, 2 * MCH, HH, WW); }

  // persistent autoregressive wavefront decode (single WGP, 16 waves, WMMA)
  size_t smem = (size_t)(16 * CS + 16 * S0 + 16 * S1) * 2 + (size_t)16 * 2 * MCH * 4;
  ar_decode_wavefront<<<1, NT, smem, stream>>>(
      fm1, w_hat, wctx, wmlp, ctx_b,
      ep_b[0], ep_b[1], ep_b[2], ep_b[3], ep_b[4], ep_b[5],
      sQLh, (float*)d_out);
}
